// OBM_class_4509715661109
// MI455X (gfx1250) — compile-verified, matrix operand order checked by reading, not run on hardware
//
#include <hip/hip_runtime.h>
#include <hip/hip_bf16.h>

// ---------------------------------------------------------------------------
// Problem constants (match reference)
// ---------------------------------------------------------------------------
#define NN 50000      // nodes
#define EE 400000     // edges
#define GG 50         // graphs
#define F_IN 16
#define F_EE 8
#define HH 128
#define HE 256
#define GFD 4
#define OUTD 8
#define EPS_MSG 1e-7f
#define EPS_BN 1e-5f
#define EPS_SM 1e-16f

typedef __bf16 bf16_t;
typedef __attribute__((ext_vector_type(16))) __bf16 v16bf;
typedef __attribute__((ext_vector_type(8)))  float  v8f;

// ---------------------------------------------------------------------------
// WMMA fragment loader.
// 16-bit A-matrix 16x32 layout (ISA 7.12.2): lane L (<16) holds row M=L,
// K = {0..7} in VGPR0..3 (2 packed per VGPR) and {16..23} in VGPR4..7;
// lanes 16..31 hold the same rows with K offset by +8.
// B is stored transposed (Ncols x Kpad), so B fragments use the same pattern
// with "row" = output column.
// ---------------------------------------------------------------------------
__device__ __forceinline__ v16bf load_frag(const bf16_t* __restrict__ base,
                                           int ld, int row, int kb, int lane) {
  const int half = (lane >> 4) & 1;
  const bf16_t* p = base + (size_t)row * ld + kb + half * 8;
  v16bf f;
#pragma unroll
  for (int v = 0; v < 4; ++v) {
    f[2 * v]         = p[2 * v];
    f[2 * v + 1]     = p[2 * v + 1];
    f[8 + 2 * v]     = p[16 + 2 * v];
    f[8 + 2 * v + 1] = p[16 + 2 * v + 1];
  }
  return f;
}

// ---------------------------------------------------------------------------
// bf16 WMMA GEMM:  C[M x Nc] = A[M x K] @ B[K x Nc] + bias (, ReLU)
// A: bf16 row-major (M x K), Bt: bf16 TRANSPOSED (Nc x K), K % 32 == 0.
// Block = 128 threads = 4 waves; wave w owns rows [blockIdx.x*64 + w*16, +16)
// and 64 columns [blockIdx.y*64, +64) => 4 WMMA tiles reusing one A fragment.
// ---------------------------------------------------------------------------
__global__ void k_gemm_bf16(const bf16_t* __restrict__ A, const bf16_t* __restrict__ Bt,
                            const float* __restrict__ bias, float* __restrict__ C,
                            int M, int K, int Ncols, int relu) {
  const int lane  = threadIdx.x & 31;
  const int wave  = threadIdx.x >> 5;
  const int mtile = blockIdx.x * 64 + wave * 16;
  const int nbase = blockIdx.y * 64;
  if (mtile >= M) return;

  v8f acc0 = {}, acc1 = {}, acc2 = {}, acc3 = {};
  const int mrow = mtile + (lane & 15);
  const int ncl  = lane & 15;

  for (int kb = 0; kb < K; kb += 32) {
    v16bf a  = load_frag(A,  K, mrow,              kb, lane);
    v16bf b0 = load_frag(Bt, K, nbase + 0*16 + ncl, kb, lane);
    v16bf b1 = load_frag(Bt, K, nbase + 1*16 + ncl, kb, lane);
    v16bf b2 = load_frag(Bt, K, nbase + 2*16 + ncl, kb, lane);
    v16bf b3 = load_frag(Bt, K, nbase + 3*16 + ncl, kb, lane);
    acc0 = __builtin_amdgcn_wmma_f32_16x16x32_bf16(false, a, false, b0, (short)0, acc0, false, false);
    acc1 = __builtin_amdgcn_wmma_f32_16x16x32_bf16(false, a, false, b1, (short)0, acc1, false, false);
    acc2 = __builtin_amdgcn_wmma_f32_16x16x32_bf16(false, a, false, b2, (short)0, acc2, false, false);
    acc3 = __builtin_amdgcn_wmma_f32_16x16x32_bf16(false, a, false, b3, (short)0, acc3, false, false);
  }

  const int half = lane >> 4;
  v8f accs[4] = {acc0, acc1, acc2, acc3};
#pragma unroll
  for (int j = 0; j < 4; ++j) {
    const int ncol = nbase + j * 16 + ncl;
    const float bv = bias[ncol];
#pragma unroll
    for (int r = 0; r < 8; ++r) {
      float v = accs[j][r] + bv;
      if (relu) v = fmaxf(v, 0.f);
      C[(size_t)(mtile + r + half * 8) * Ncols + ncol] = v;
    }
  }
}

// ---------------------------------------------------------------------------
// Utility / conversion kernels
// ---------------------------------------------------------------------------
__global__ void k_zero_f32(float* __restrict__ p, long long n) {
  long long t = (long long)blockIdx.x * blockDim.x + threadIdx.x;
  if (t < n) p[t] = 0.f;
}

// X[M x K] fp32 -> Xb[M x Kpad] bf16, zero padded in K
__global__ void k_cvt_rows_pad(const float* __restrict__ X, bf16_t* __restrict__ Xb,
                               int M, int K, int Kpad) {
  long long t = (long long)blockIdx.x * blockDim.x + threadIdx.x;
  if (t >= (long long)M * Kpad) return;
  int k = (int)(t % Kpad);
  int m = (int)(t / Kpad);
  Xb[t] = (k < K) ? (bf16_t)X[(size_t)m * K + k] : (bf16_t)0.f;
}

// W[K x N] fp32 -> Wt[N x Kpad] bf16 (transposed, zero padded in K)
__global__ void k_cvt_wT_pad(const float* __restrict__ W, bf16_t* __restrict__ Wt,
                             int K, int Ncols, int Kpad) {
  long long t = (long long)blockIdx.x * blockDim.x + threadIdx.x;
  if (t >= (long long)Ncols * Kpad) return;
  int k = (int)(t % Kpad);
  int n = (int)(t / Kpad);
  Wt[t] = (k < K) ? (bf16_t)W[(size_t)k * Ncols + n] : (bf16_t)0.f;
}

// ---------------------------------------------------------------------------
// Edge passes: msg = relu(hs[src] + edge_attr@We + be) + eps  (recomputed)
// Pass 1: per-dst per-feature max via int-bits atomicMax (msg > 0, init 0).
// Pass 2: ex = exp(msg - mx[dst]); den += ex; num += ex*msg (float atomics).
// Thread mapping: tid = e*128 + f -> coalesced 512B gathers of hs[src[e]].
// ---------------------------------------------------------------------------
__global__ void k_edge_max(const float* __restrict__ hs, const float* __restrict__ ea,
                           const float* __restrict__ We, const float* __restrict__ be,
                           const int* __restrict__ src, const int* __restrict__ dst,
                           int* __restrict__ mx) {
  __shared__ float sWe[F_EE * HH];
  __shared__ float sbe[HH];
  for (int i = threadIdx.x; i < F_EE * HH; i += blockDim.x) sWe[i] = We[i];
  for (int i = threadIdx.x; i < HH; i += blockDim.x) sbe[i] = be[i];
  __syncthreads();
  long long t = (long long)blockIdx.x * blockDim.x + threadIdx.x;
  if (t >= (long long)EE * HH) return;
  int e = (int)(t >> 7), f = (int)(t & 127);
  float acc = sbe[f];
#pragma unroll
  for (int k = 0; k < F_EE; ++k) acc += ea[(size_t)e * F_EE + k] * sWe[k * HH + f];
  float msg = fmaxf(hs[(size_t)src[e] * HH + f] + acc, 0.f) + EPS_MSG;
  atomicMax(&mx[(size_t)dst[e] * HH + f], __float_as_int(msg));
}

__global__ void k_edge_sum(const float* __restrict__ hs, const float* __restrict__ ea,
                           const float* __restrict__ We, const float* __restrict__ be,
                           const int* __restrict__ src, const int* __restrict__ dst,
                           const float* __restrict__ mx, float* __restrict__ den,
                           float* __restrict__ num) {
  __shared__ float sWe[F_EE * HH];
  __shared__ float sbe[HH];
  for (int i = threadIdx.x; i < F_EE * HH; i += blockDim.x) sWe[i] = We[i];
  for (int i = threadIdx.x; i < HH; i += blockDim.x) sbe[i] = be[i];
  __syncthreads();
  long long t = (long long)blockIdx.x * blockDim.x + threadIdx.x;
  if (t >= (long long)EE * HH) return;
  int e = (int)(t >> 7), f = (int)(t & 127);
  float acc = sbe[f];
#pragma unroll
  for (int k = 0; k < F_EE; ++k) acc += ea[(size_t)e * F_EE + k] * sWe[k * HH + f];
  float msg = fmaxf(hs[(size_t)src[e] * HH + f] + acc, 0.f) + EPS_MSG;
  size_t di = (size_t)dst[e] * HH + f;
  float ex = __expf(msg - mx[di]);
  atomicAdd(&den[di], ex);
  atomicAdd(&num[di], ex * msg);
}

// agg = num/(den+eps) + x_dst  -> bf16 (feeds WMMA GEMM1)
__global__ void k_node_update(const float* __restrict__ num, const float* __restrict__ den,
                              const float* __restrict__ xd, bf16_t* __restrict__ aggb) {
  long long t = (long long)blockIdx.x * blockDim.x + threadIdx.x;
  if (t >= (long long)NN * HH) return;
  float v = num[t] / (den[t] + EPS_SM) + xd[t];
  aggb[t] = (bf16_t)v;
}

// ---------------------------------------------------------------------------
// BatchNorm over batch (population stats), ReLU, -> bf16
// ---------------------------------------------------------------------------
__global__ void k_bn_partial(const float* __restrict__ hid, float* __restrict__ sums,
                             float* __restrict__ sumsq, int M) {
  int f = threadIdx.x;  // blockDim.x == HE
  int rows = (M + gridDim.x - 1) / gridDim.x;
  int r0 = blockIdx.x * rows;
  int r1 = min(M, r0 + rows);
  float s = 0.f, sq = 0.f;
  for (int r = r0; r < r1; ++r) {
    float v = hid[(size_t)r * HE + f];
    s += v; sq += v * v;
  }
  atomicAdd(&sums[f], s);
  atomicAdd(&sumsq[f], sq);
}

__global__ void k_bn_apply(const float* __restrict__ hid, const float* __restrict__ sums,
                           const float* __restrict__ sumsq, const float* __restrict__ gam,
                           const float* __restrict__ bet, bf16_t* __restrict__ outb, int M) {
  long long t = (long long)blockIdx.x * blockDim.x + threadIdx.x;
  if (t >= (long long)M * HE) return;
  int f = (int)(t % HE);
  float inv = 1.f / (float)M;
  float mu = sums[f] * inv;
  float var = sumsq[f] * inv - mu * mu;
  float y = (hid[t] - mu) * rsqrtf(var + EPS_BN) * gam[f] + bet[f];
  outb[t] = (bf16_t)fmaxf(y, 0.f);
}

// ---------------------------------------------------------------------------
// Global mean pool + head (softmax)
// ---------------------------------------------------------------------------
__global__ void k_pool(const float* __restrict__ h, const int* __restrict__ batch,
                       float* __restrict__ pooled, float* __restrict__ cnt) {
  long long t = (long long)blockIdx.x * blockDim.x + threadIdx.x;
  if (t >= (long long)NN * HH) return;
  int n = (int)(t >> 7), f = (int)(t & 127);
  int b = batch[n];
  atomicAdd(&pooled[(size_t)b * HH + f], h[t]);
  if (f == 0) atomicAdd(&cnt[b], 1.0f);
}

__global__ void k_head(const float* __restrict__ pooled, const float* __restrict__ cnt,
                       const float* __restrict__ gfeat, const float* __restrict__ Wh,
                       const float* __restrict__ bh, float* __restrict__ out) {
  __shared__ float z[HH + GFD];
  __shared__ float logits[OUTD];
  int g = blockIdx.x, lane = threadIdx.x;
  float ic = 1.f / cnt[g];
  for (int i = lane; i < HH; i += 32) z[i] = pooled[(size_t)g * HH + i] * ic;
  if (lane < GFD) z[HH + lane] = gfeat[g * GFD + lane];
  __syncthreads();
  if (lane < OUTD) {
    float acc = bh[lane];
    for (int k = 0; k < HH + GFD; ++k) acc += z[k] * Wh[k * OUTD + lane];
    logits[lane] = acc;
  }
  __syncthreads();
  if (lane < OUTD) {
    float m = logits[0];
    for (int j = 1; j < OUTD; ++j) m = fmaxf(m, logits[j]);
    float s = 0.f;
    for (int j = 0; j < OUTD; ++j) s += __expf(logits[j] - m);
    out[(size_t)g * OUTD + lane] = __expf(logits[lane] - m) / s;
  }
}

// ---------------------------------------------------------------------------
// Host orchestration
// ---------------------------------------------------------------------------
static inline unsigned nblk(long long total, int bs) {
  return (unsigned)((total + bs - 1) / bs);
}

extern "C" void kernel_launch(void* const* d_in, const int* in_sizes, int n_in,
                              void* d_out, int out_size, void* d_ws, size_t ws_size,
                              hipStream_t stream) {
  const float* x      = (const float*)d_in[0];
  const float* ea     = (const float*)d_in[1];
  const float* gfeat  = (const float*)d_in[2];
  const float* W_src0 = (const float*)d_in[3];
  const float* b_src0 = (const float*)d_in[4];
  const float* W_dst0 = (const float*)d_in[5];
  const float* b_dst0 = (const float*)d_in[6];
  const float* W_e0   = (const float*)d_in[7];
  const float* b_e0   = (const float*)d_in[8];
  const float* Wm1_0  = (const float*)d_in[9];
  const float* bm1_0  = (const float*)d_in[10];
  const float* g0     = (const float*)d_in[11];
  const float* beta0  = (const float*)d_in[12];
  const float* Wm2_0  = (const float*)d_in[13];
  const float* bm2_0  = (const float*)d_in[14];
  const float* W_e    = (const float*)d_in[15];
  const float* b_e    = (const float*)d_in[16];
  const float* Wm1    = (const float*)d_in[17];
  const float* bm1    = (const float*)d_in[18];
  const float* gam    = (const float*)d_in[19];
  const float* bet    = (const float*)d_in[20];
  const float* Wm2    = (const float*)d_in[21];
  const float* bm2    = (const float*)d_in[22];
  const float* W_head = (const float*)d_in[23];
  const float* b_head = (const float*)d_in[24];
  const int* edge_index = (const int*)d_in[25];
  const int* batch_ids  = (const int*)d_in[26];
  const int* src = edge_index;
  const int* dst = edge_index + EE;
  float* out = (float*)d_out;
  (void)in_sizes; (void)n_in; (void)out_size; (void)ws_size;

  // ---- workspace carving (256B aligned) ----
  char* p = (char*)d_ws;
  auto carve = [&](size_t bytes) -> void* {
    void* r = (void*)p;
    p += (bytes + 255) & ~(size_t)255;
    return r;
  };
  float*  h_src  = (float*)carve((size_t)NN * HH * 4);
  float*  h_dst  = (float*)carve((size_t)NN * HH * 4);
  float*  h      = (float*)carve((size_t)NN * HH * 4);
  float*  mx     = (float*)carve((size_t)NN * HH * 4);  // also viewed as int*
  float*  den    = (float*)carve((size_t)NN * HH * 4);
  float*  num    = (float*)carve((size_t)NN * HH * 4);
  float*  hid    = (float*)carve((size_t)NN * HE * 4);
  bf16_t* aggb   = (bf16_t*)carve((size_t)NN * HH * 2);
  bf16_t* hidb   = (bf16_t*)carve((size_t)NN * HE * 2);
  bf16_t* xb     = (bf16_t*)carve((size_t)NN * 32 * 2);
  bf16_t* WsT    = (bf16_t*)carve((size_t)HH * 32 * 2);
  bf16_t* WdT    = (bf16_t*)carve((size_t)HH * 32 * 2);
  bf16_t* W1T    = (bf16_t*)carve((size_t)HE * HH * 2);
  bf16_t* W2T    = (bf16_t*)carve((size_t)HH * HE * 2);
  float*  bnsum  = (float*)carve(HE * 4);
  float*  bnsq   = (float*)carve(HE * 4);
  float*  pooled = (float*)carve((size_t)GG * HH * 4);
  float*  cnt    = (float*)carve((size_t)GG * 4);

  const long long NH = (long long)NN * HH;
  const long long EH = (long long)EE * HH;

  // ---- layer 0 projections (K=16 padded to 32) ----
  k_cvt_rows_pad<<<nblk((long long)NN * 32, 256), 256, 0, stream>>>(x, xb, NN, F_IN, 32);
  k_cvt_wT_pad<<<nblk((long long)HH * 32, 256), 256, 0, stream>>>(W_src0, WsT, F_IN, HH, 32);
  k_cvt_wT_pad<<<nblk((long long)HH * 32, 256), 256, 0, stream>>>(W_dst0, WdT, F_IN, HH, 32);
  {
    dim3 gs((NN + 63) / 64, HH / 64);
    k_gemm_bf16<<<gs, 128, 0, stream>>>(xb, WsT, b_src0, h_src, NN, 32, HH, 0);
    k_gemm_bf16<<<gs, 128, 0, stream>>>(xb, WdT, b_dst0, h_dst, NN, 32, HH, 0);
  }

  for (int layer = 0; layer < 3; ++layer) {
    const float* hs  = (layer == 0) ? h_src : h;
    const float* xd  = (layer == 0) ? h_dst : h;
    const float* Wel = (layer == 0) ? W_e0  : W_e  + (size_t)(layer - 1) * F_EE * HH;
    const float* bel = (layer == 0) ? b_e0  : b_e  + (size_t)(layer - 1) * HH;
    const float* W1l = (layer == 0) ? Wm1_0 : Wm1  + (size_t)(layer - 1) * HH * HE;
    const float* b1l = (layer == 0) ? bm1_0 : bm1  + (size_t)(layer - 1) * HE;
    const float* gl  = (layer == 0) ? g0    : gam  + (size_t)(layer - 1) * HE;
    const float* btl = (layer == 0) ? beta0 : bet  + (size_t)(layer - 1) * HE;
    const float* W2l = (layer == 0) ? Wm2_0 : Wm2  + (size_t)(layer - 1) * HE * HH;
    const float* b2l = (layer == 0) ? bm2_0 : bm2  + (size_t)(layer - 1) * HH;

    // reset aggregation buffers
    k_zero_f32<<<nblk(NH, 256), 256, 0, stream>>>(mx, NH);
    k_zero_f32<<<nblk(NH, 256), 256, 0, stream>>>(den, NH);
    k_zero_f32<<<nblk(NH, 256), 256, 0, stream>>>(num, NH);

    // softmax aggregation over edges (2 passes, inline edge embedding)
    k_edge_max<<<nblk(EH, 256), 256, 0, stream>>>(hs, ea, Wel, bel, src, dst, (int*)mx);
    k_edge_sum<<<nblk(EH, 256), 256, 0, stream>>>(hs, ea, Wel, bel, src, dst, mx, den, num);
    k_node_update<<<nblk(NH, 256), 256, 0, stream>>>(num, den, xd, aggb);

    // MLP: Linear(128->256) -> BN -> ReLU -> Linear(256->128) -> ReLU
    k_cvt_wT_pad<<<nblk((long long)HE * HH, 256), 256, 0, stream>>>(W1l, W1T, HH, HE, HH);
    {
      dim3 gs((NN + 63) / 64, HE / 64);
      k_gemm_bf16<<<gs, 128, 0, stream>>>(aggb, W1T, b1l, hid, NN, HH, HE, 0);
    }
    k_zero_f32<<<nblk(HE, 256), 256, 0, stream>>>(bnsum, HE);
    k_zero_f32<<<nblk(HE, 256), 256, 0, stream>>>(bnsq, HE);
    k_bn_partial<<<512, HE, 0, stream>>>(hid, bnsum, bnsq, NN);
    k_bn_apply<<<nblk((long long)NN * HE, 256), 256, 0, stream>>>(hid, bnsum, bnsq, gl, btl, hidb, NN);
    k_cvt_wT_pad<<<nblk((long long)HH * HE, 256), 256, 0, stream>>>(W2l, W2T, HE, HH, HE);
    {
      dim3 gs((NN + 63) / 64, HH / 64);
      k_gemm_bf16<<<gs, 128, 0, stream>>>(hidb, W2T, b2l, h, NN, HE, HH, 1); // fused final ReLU
    }
  }

  // global mean pool + head softmax
  k_zero_f32<<<nblk((long long)GG * HH, 256), 256, 0, stream>>>(pooled, (long long)GG * HH);
  k_zero_f32<<<1, 256, 0, stream>>>(cnt, GG);
  k_pool<<<nblk(NH, 256), 256, 0, stream>>>(h, batch_ids, pooled, cnt);
  k_head<<<GG, 32, 0, stream>>>(pooled, cnt, gfeat, W_head, b_head, out);
}
